// MultiHeadAttention_36172214567417
// MI455X (gfx1250) — compile-verified, matrix operand order checked by reading, not run on hardware
//
#include <hip/hip_runtime.h>

// ---------------------------------------------------------------------------
// MI455X (gfx1250) fused multi-head attention, bf16 WMMA pipeline.
// N=4, L=S=2048, D=1024, H=16, DK=64. Compute-bound (~137 GFLOP vs ~250MB
// traffic) -> all GEMMs on v_wmma_f32_16x16x32_bf16. The reference softmax is
// over the batch axis N (4 values): purely elementwise, exchanged through LDS
// so each wave owns exactly ONE batch. S streamed in 64-wide tiles; the
// cross-batch exchange uses a DS-only barrier (s_wait_dscnt + split barrier)
// so global loads/prefetches keep flowing across it.
// ---------------------------------------------------------------------------

typedef __attribute__((ext_vector_type(16))) __bf16 v16bf;
typedef __attribute__((ext_vector_type(8)))  __bf16 v8bf;
typedef __attribute__((ext_vector_type(4)))  __bf16 v4bf;
typedef __attribute__((ext_vector_type(8)))  float  v8f;
typedef __attribute__((ext_vector_type(4)))  float  v4f;

#define NB   4
#define LQ   2048
#define SK   2048
#define DIM  1024
#define HH   16
#define DK   64
#define ROWS (NB*LQ)   // 8192 flattened (n,l) rows

__device__ __forceinline__ v8f wmma_bf16(v16bf a, v16bf b, v8f c) {
  return __builtin_amdgcn_wmma_f32_16x16x32_bf16(false, a, false, b,
                                                 (short)0, c, false, false);
}

// Workgroup barrier that only orders LDS: makes this wave's ds_stores visible
// (dscnt==0) then split-signals/waits the WG barrier. Unlike __syncthreads()
// it does NOT drain loadcnt/storecnt, so global loads and prefetches issued
// before it stay in flight across the barrier.
__device__ __forceinline__ void ds_barrier() {
  asm volatile("s_wait_dscnt 0x0\n\t"
               "s_barrier_signal -1\n\t"
               "s_barrier_wait -1" ::: "memory");
}

// Load a 16x32 bf16 fragment from a row-major matrix (used for BOTH the A
// operand (rows = M) and the B operand loaded from its row-major (N x K)
// source). CDNA5 16-bit A layout: lanes 0-15 hold row = lane,
// K = [0..7, 16..23]; lanes 16-31 hold row = lane-16, K = [8..15, 24..31].
__device__ __forceinline__ v16bf load_frag(const __bf16* __restrict__ base,
                                           int ld, int row0, int k0) {
  const int lane = threadIdx.x & 31;
  const int row  = row0 + (lane & 15);
  const int kb   = k0 + ((lane >> 4) << 3);
  const __bf16* p = base + (size_t)row * ld + kb;
  v8bf lo = *(const v8bf*)(p);        // K = kb .. kb+7
  v8bf hi = *(const v8bf*)(p + 16);   // K = kb+16 .. kb+23
  v16bf r;
#pragma unroll
  for (int i = 0; i < 8; ++i) { r[i] = lo[i]; r[i + 8] = hi[i]; }
  return r;
}

// Same fragment layout, sourced from an LDS tile with row stride ldsw.
__device__ __forceinline__ v16bf load_frag_lds(const __bf16* t, int ldsw) {
  const int lane = threadIdx.x & 31;
  const int row  = lane & 15;
  const int kb   = (lane >> 4) << 3;
  v8bf lo = *(const v8bf*)(t + row * ldsw + kb);
  v8bf hi = *(const v8bf*)(t + row * ldsw + kb + 16);
  v16bf r;
#pragma unroll
  for (int i = 0; i < 8; ++i) { r[i] = lo[i]; r[i + 8] = hi[i]; }
  return r;
}

// ------------------------------ f32 -> bf16 --------------------------------
__global__ void cvt_bf16_kernel(const float* __restrict__ src,
                                __bf16* __restrict__ dst) {
  size_t i = ((size_t)blockIdx.x * blockDim.x + threadIdx.x) * 4;
  v4f f = *(const v4f*)(src + i);
  v4bf b;
#pragma unroll
  for (int j = 0; j < 4; ++j) b[j] = (__bf16)f[j];
  *(v4bf*)(dst + i) = b;
}

// --------------------- per-head projection GEMM ----------------------------
// C[h] = A(8192x1024) x Wq[h]^T(1024x64), written bf16.
// MODE 0: row-major [h][row][c] with scale (q gets 1/sqrt(dk)).
// MODE 1: transposed per batch: vT[h][n][c][s]  (so attn@v B-frags are
//         contiguous rows later, no strided loads).
template <int MODE>
__global__ void proj_kernel(const __bf16* __restrict__ A,
                            const __bf16* __restrict__ Wq,
                            __bf16* __restrict__ out, float scale) {
  const int wave = threadIdx.x >> 5;
  const int h    = blockIdx.y;
  const int row0 = blockIdx.x * 128 + wave * 16;
  const __bf16* B = Wq + (size_t)h * DK * DIM;

  v8f acc[4] = {};
  for (int k = 0; k < DIM; k += 32) {
    __builtin_prefetch(A + (size_t)row0 * DIM + k + 128, 0, 0);
    v16bf a = load_frag(A, DIM, row0, k);
#pragma unroll
    for (int t = 0; t < 4; ++t) {
      v16bf b = load_frag(B, DIM, t * 16, k);
      acc[t] = wmma_bf16(a, b, acc[t]);
    }
  }

  const int lane = threadIdx.x & 31;
  const int col  = lane & 15;
  const int hi   = lane >> 4;
  if (MODE == 0) {
    __bf16* o = out + (size_t)h * ROWS * DK;
#pragma unroll
    for (int t = 0; t < 4; ++t)
#pragma unroll
      for (int r = 0; r < 8; ++r) {
        int row = row0 + r + 8 * hi;
        o[(size_t)row * DK + t * 16 + col] = (__bf16)(acc[t][r] * scale);
      }
  } else {
    __bf16* o = out + (size_t)h * NB * DK * SK;
#pragma unroll
    for (int t = 0; t < 4; ++t)
#pragma unroll
      for (int r = 0; r < 8; ++r) {
        int row = row0 + r + 8 * hi;        // = n*2048 + s
        int n = row >> 11, s = row & 2047;
        int c = t * 16 + col;
        o[((size_t)n * DK + c) * SK + s] = (__bf16)(acc[t][r] * scale);
      }
  }
}

// ----------------------- fused attention kernel ----------------------------
// grid(L/32, H), 256 threads. Wave (n, lh) owns batch n and 16 L-rows;
// streams S in 64-wide tiles. Cross-batch softmax exchanged via LDS
// (double-buffered, one DS-only barrier per S-tile).
__global__ void attn_kernel(const __bf16* __restrict__ qp,
                            const __bf16* __restrict__ kp,
                            const __bf16* __restrict__ vt,
                            __bf16* __restrict__ Hd) {
  constexpr int LDSW = 72;                       // 64 cols + pad
  // raw-score exchange: [buf][lh][n][ct][lane][r]  (per-lane contiguous r)
  __shared__ __align__(16) float sbuf[2][2][NB][4][32][8];   // 64 KB
  // per-wave attn tile for C-frag -> A-frag relayout (16 x 64 bf16)
  __shared__ __align__(16) __bf16 atile[8][16 * LDSW + 8];   // ~18 KB

  const int wave = threadIdx.x >> 5;
  const int lane = threadIdx.x & 31;
  const int col  = lane & 15;
  const int hi   = lane >> 4;
  const int n    = wave >> 1;                    // batch owned by this wave
  const int lh   = wave & 1;                     // L-subtile within WG
  const int h    = blockIdx.y;
  const int l0   = blockIdx.x * 32 + lh * 16;

  const __bf16* qh  = qp + (size_t)h * ROWS * DK;
  const __bf16* kh  = kp + (size_t)h * ROWS * DK;
  const __bf16* vhn = vt + (size_t)h * NB * DK * SK + (size_t)n * DK * SK;
  __bf16* myt = &atile[wave][0];
  const int n1 = (n + 1) & 3, n2 = (n + 2) & 3, n3 = (n + 3) & 3;

  // resident q fragments for this wave's batch (K=64 -> 2 frags)
  v16bf aq[2];
#pragma unroll
  for (int kk = 0; kk < 2; ++kk)
    aq[kk] = load_frag(qh, DK, n * LQ + l0, kk * 32);

  v8f acc[4] = {};                               // 16 x 64 f32, one batch

  for (int s0 = 0; s0 < SK; s0 += 64) {
    const int buf = (s0 >> 6) & 1;

    // ---- scores for own batch: q . k^T (q pre-scaled by 1/8) ----
    v8f sc[4] = {};
#pragma unroll
    for (int kk = 0; kk < 2; ++kk)
#pragma unroll
      for (int ct = 0; ct < 4; ++ct) {
        v16bf b = load_frag(kh, DK, n * LQ + s0 + ct * 16, kk * 32);
        sc[ct] = wmma_bf16(aq[kk], b, sc[ct]);
      }

    // ---- publish raw scores (per-lane contiguous -> b128 stores) ----
    {
      float* myb = &sbuf[buf][lh][n][0][lane][0];
#pragma unroll
      for (int ct = 0; ct < 4; ++ct)
#pragma unroll
        for (int r = 0; r < 8; ++r)
          myb[ct * 32 * 8 + r] = sc[ct][r];
    }

    // ---- prefetch this tile's v BEFORE the barrier: no VGPR / no counter
    //      cost; the lines land in L0/L2 while we synchronize + softmax.
    //      Rows = dk (64 of them), each needs bytes [2*s0, 2*s0+128). ----
    {
      const __bf16* pv = vhn + (size_t)lane * SK + s0;
      __builtin_prefetch(pv, 0, 3);                  // rows 0..31, first line
      __builtin_prefetch(pv + 32, 0, 3);             // rows 0..31, second line
      __builtin_prefetch(pv + (size_t)32 * SK, 0, 3);      // rows 32..63
      __builtin_prefetch(pv + (size_t)32 * SK + 32, 0, 3);
    }

    ds_barrier();   // DS-only: prefetches/loads stay in flight

    // ---- cross-batch softmax: read 3 partners, keep only own numerator ----
#pragma unroll
    for (int ct = 0; ct < 4; ++ct) {
      const float* b1 = &sbuf[buf][lh][n1][ct][lane][0];
      const float* b2 = &sbuf[buf][lh][n2][ct][lane][0];
      const float* b3 = &sbuf[buf][lh][n3][ct][lane][0];
#pragma unroll
      for (int r = 0; r < 8; ++r) {
        float s0v = sc[ct][r], s1 = b1[r], s2 = b2[r], s3 = b3[r];
        float m  = fmaxf(fmaxf(s0v, s1), fmaxf(s2, s3));
        float e0 = __expf(s0v - m);
        float den = e0 + __expf(s1 - m) + __expf(s2 - m) + __expf(s3 - m);
        sc[ct][r] = e0 * (1.0f / den);
      }
    }

    // ---- attn @ v : wave-private LDS relayout (C-frag -> A-frag) ----
#pragma unroll
    for (int ct = 0; ct < 4; ++ct)
#pragma unroll
      for (int r = 0; r < 8; ++r)
        myt[(r + 8 * hi) * LDSW + ct * 16 + col] = (__bf16)sc[ct][r];
    // wave-private buffer; LDS is in-order within a wave -- just stop the
    // compiler from reordering the ds_load above the ds_stores.
    asm volatile("" ::: "memory");
    v16bf ap0 = load_frag_lds(myt, LDSW);        // attn K = s0   .. s0+31
    v16bf ap1 = load_frag_lds(myt + 32, LDSW);   // attn K = s0+32.. s0+63
    asm volatile("" ::: "memory");
#pragma unroll
    for (int t = 0; t < 4; ++t) {
      v16bf bv = load_frag(vhn, SK, t * 16, s0);        // prefetched -> hits
      acc[t] = wmma_bf16(ap0, bv, acc[t]);
    }
#pragma unroll
    for (int t = 0; t < 4; ++t) {
      v16bf bv = load_frag(vhn, SK, t * 16, s0 + 32);
      acc[t] = wmma_bf16(ap1, bv, acc[t]);
    }
    // NOTE: one barrier per iteration suffices: sbuf is double-buffered, so
    // iteration i+2's writes to the same buffer sit behind barrier i+1, which
    // this wave only signals after its reads of buffer (i&1) completed.
  }

  // ---- write heads, already concatenated: Hd[n][l][h*64 + c] (bf16) ----
#pragma unroll
  for (int t = 0; t < 4; ++t)
#pragma unroll
    for (int r = 0; r < 8; ++r) {
      int l = l0 + r + 8 * hi;
      Hd[((size_t)n * LQ + l) * (HH * DK) + h * DK + t * 16 + col] =
          (__bf16)acc[t][r];
    }
}

// --------------------------- output projection -----------------------------
// out(8192x1024) = Hd(8192x1024) x Wo^T(1024x1024), f32 result.
__global__ void outproj_kernel(const __bf16* __restrict__ A,
                               const __bf16* __restrict__ Wo,
                               float* __restrict__ out) {
  const int wave = threadIdx.x >> 5;
  const int row0 = blockIdx.x * 128 + wave * 16;
  const int col0 = blockIdx.y * 64;
  const __bf16* B = Wo + (size_t)col0 * DIM;

  v8f acc[4] = {};
  for (int k = 0; k < DIM; k += 32) {
    __builtin_prefetch(A + (size_t)row0 * DIM + k + 128, 0, 0);
    v16bf a = load_frag(A, DIM, row0, k);
#pragma unroll
    for (int t = 0; t < 4; ++t) {
      v16bf b = load_frag(B, DIM, t * 16, k);
      acc[t] = wmma_bf16(a, b, acc[t]);
    }
  }

  const int lane = threadIdx.x & 31;
  const int col  = lane & 15;
  const int hi   = lane >> 4;
#pragma unroll
  for (int t = 0; t < 4; ++t)
#pragma unroll
    for (int r = 0; r < 8; ++r) {
      int row = row0 + r + 8 * hi;
      out[(size_t)row * DIM + col0 + t * 16 + col] = acc[t][r];
    }
}

// ---------------------------------------------------------------------------
extern "C" void kernel_launch(void* const* d_in, const int* in_sizes, int n_in,
                              void* d_out, int out_size, void* d_ws, size_t ws_size,
                              hipStream_t stream) {
  const float* Q  = (const float*)d_in[0];   // (N,L,D)
  const float* K  = (const float*)d_in[1];   // (N,S,D)
  const float* V  = (const float*)d_in[2];   // (N,S,D)
  const float* Wq = (const float*)d_in[3];   // (H,DK,D)
  const float* Wo = (const float*)d_in[4];   // (D, H*DK)
  float* out = (float*)d_out;

  const size_t SZ_IN = (size_t)NB * LQ * DIM;     // 8,388,608
  const size_t SZ_W  = (size_t)HH * DK * DIM;     // 1,048,576
  const size_t SZ_P  = (size_t)HH * ROWS * DK;    // 8,388,608

  char* ws = (char*)d_ws;
  size_t off = 0;
  auto carve = [&](size_t bytes) {
    void* p = ws + off;
    off += (bytes + 255) & ~(size_t)255;
    return p;
  };
  __bf16* Qb  = (__bf16*)carve(SZ_IN * 2);
  __bf16* Kb  = (__bf16*)carve(SZ_IN * 2);
  __bf16* Vb  = (__bf16*)carve(SZ_IN * 2);
  __bf16* Wqb = (__bf16*)carve(SZ_W * 2);
  __bf16* Wob = (__bf16*)carve(SZ_W * 2);
  __bf16* qpj = (__bf16*)carve(SZ_P * 2);   // [h][n*L+l][dk]
  __bf16* kpj = (__bf16*)carve(SZ_P * 2);   // [h][n*S+s][dk]
  __bf16* vtp = (__bf16*)carve(SZ_P * 2);   // [h][n][dk][s]  (transposed)
  __bf16* Hd  = (__bf16*)carve(SZ_P * 2);   // [n][l][h*dk+c]

  // 1) convert everything to bf16 once (whole ~116MB working set then stays
  //    resident in the 192MB L2).
  cvt_bf16_kernel<<<(int)(SZ_IN / 1024), 256, 0, stream>>>(Q, Qb);
  cvt_bf16_kernel<<<(int)(SZ_IN / 1024), 256, 0, stream>>>(K, Kb);
  cvt_bf16_kernel<<<(int)(SZ_IN / 1024), 256, 0, stream>>>(V, Vb);
  cvt_bf16_kernel<<<(int)(SZ_W / 1024), 256, 0, stream>>>(Wq, Wqb);
  cvt_bf16_kernel<<<(int)(SZ_W / 1024), 256, 0, stream>>>(Wo, Wob);

  // 2) projections (all three use Wq, faithful to the reference).
  //    q is pre-scaled by 1/sqrt(dk) = 0.125 so scores need no extra scale.
  dim3 pg(ROWS / 128, HH);
  proj_kernel<0><<<pg, 256, 0, stream>>>(Qb, Wqb, qpj, 0.125f);
  proj_kernel<0><<<pg, 256, 0, stream>>>(Kb, Wqb, kpj, 1.0f);
  proj_kernel<1><<<pg, 256, 0, stream>>>(Vb, Wqb, vtp, 1.0f);

  // 3) fused scores -> cross-batch softmax -> attn@v -> concat heads.
  attn_kernel<<<dim3(LQ / 32, HH), 256, 0, stream>>>(qpj, kpj, vtp, Hd);

  // 4) output projection, f32 result.
  outproj_kernel<<<dim3(ROWS / 128, DIM / 64), 256, 0, stream>>>(Hd, Wob, out);
  (void)in_sizes; (void)n_in; (void)out_size; (void)ws_size;
}
